// Attention_Weight_79370995630410
// MI455X (gfx1250) — matrix operand order repaired and checked by reading, not compiled, not run on hardware
//
#include <hip/hip_runtime.h>
#include <hip/hip_bf16.h>

typedef _Float16 half_t;
typedef __attribute__((ext_vector_type(16))) _Float16 v16h;
typedef __attribute__((ext_vector_type(8)))  _Float16 v8h;
typedef __attribute__((ext_vector_type(4)))  _Float16 v4h;
typedef __attribute__((ext_vector_type(8)))  float    v8f;
typedef __attribute__((ext_vector_type(4)))  float    v4f;

constexpr int kB = 8;
constexpr int kS = 2048;
constexpr int kD = 1024;

#define WMMA_F16(a, b, c) \
  __builtin_amdgcn_wmma_f32_16x16x32_f16(false, (a), false, (b), (short)0, (c), false, false)

// A-fragment (16x32 f16, row-major source): lane l -> row (l&15),
// halves k0+(l>=16?8:0)+[0..7] and +16..23  => two b128 loads.
__device__ __forceinline__ v16h frag_a(const half_t* base, int ld, int row0, int k0, int lane) {
  const half_t* p = base + (size_t)(row0 + (lane & 15)) * ld + k0 + ((lane >> 4) << 3);
  union { v16h v; v8h h[2]; } u;
  u.h[0] = *(const v8h*)(p);
  u.h[1] = *(const v8h*)(p + 16);
  return u.v;
}

// B-fragment (32x16 f16) from B^T row-major storage (rows = N columns of B):
// lane l -> column n0+(l&15), halves k0+(l>=16?8:0)+[0..7] and +16..23 => two b128 loads.
__device__ __forceinline__ v16h frag_bt(const half_t* base, int ld, int n0, int k0, int lane) {
  const half_t* p = base + (size_t)(n0 + (lane & 15)) * ld + k0 + ((lane >> 4) << 3);
  union { v16h v; v8h h[2]; } u;
  u.h[0] = *(const v8h*)(p);
  u.h[1] = *(const v8h*)(p + 16);
  return u.v;
}

// ---------------------------------------------------------------- cvt_x
__global__ __launch_bounds__(256) void cvt_x_kernel(const float* __restrict__ x,
                                                    half_t* __restrict__ xh) {
  int i = (blockIdx.x * 256 + threadIdx.x) * 4;   // grid sized exactly
  v4f f = *(const v4f*)(x + i);
  v4h h;
  h[0] = (half_t)f[0]; h[1] = (half_t)f[1]; h[2] = (half_t)f[2]; h[3] = (half_t)f[3];
  *(v4h*)(xh + i) = h;
}

// ------------------------------------------------- cvt_w: f32 -> f16, transposed
__global__ __launch_bounds__(256) void cvt_w_kernel(const float* __restrict__ Wq,
                                                    const float* __restrict__ Wk,
                                                    const float* __restrict__ Wv,
                                                    half_t* __restrict__ wT) {
  const float* W = (blockIdx.z == 0) ? Wq : ((blockIdx.z == 1) ? Wk : Wv);
  half_t* dst = wT + (size_t)blockIdx.z * kD * kD;
  __shared__ float tile[32][33];
  int tx = threadIdx.x & 31, ty = threadIdx.x >> 5;
  int kbase = blockIdx.y * 32, nbase = blockIdx.x * 32;
#pragma unroll
  for (int j = 0; j < 4; ++j)
    tile[ty + j * 8][tx] = W[(size_t)(kbase + ty + j * 8) * kD + nbase + tx];
  __syncthreads();
#pragma unroll
  for (int j = 0; j < 4; ++j)
    dst[(size_t)(nbase + ty + j * 8) * kD + kbase + tx] = (half_t)tile[tx][ty + j * 8];
}

// ---------------------------------------------------------------- QKV GEMM
// Y = x @ W + b ; workgroup tile 128x128, 8 waves (4 M x 2 N), wave tile 32x64.
__global__ __launch_bounds__(256) void qkv_kernel(const half_t* __restrict__ xh,
                                                  const half_t* __restrict__ wT,
                                                  const float* __restrict__ bq,
                                                  const float* __restrict__ bk,
                                                  const float* __restrict__ bv,
                                                  half_t* __restrict__ qh,
                                                  half_t* __restrict__ kh,
                                                  half_t* __restrict__ vT) {
  const int p = blockIdx.z;
  const half_t* w = wT + (size_t)p * kD * kD;
  const float* bias = (p == 0) ? bq : ((p == 1) ? bk : bv);
  const int lane = threadIdx.x & 31, wid = threadIdx.x >> 5;
  const int m0 = blockIdx.x * 128 + (wid >> 1) * 32;
  const int n0 = blockIdx.y * 128 + (wid & 1) * 64;

  v8f acc[2][4] = {};
  for (int k0 = 0; k0 < kD; k0 += 32) {
    v16h a0 = frag_a(xh, kD, m0,      k0, lane);
    v16h a1 = frag_a(xh, kD, m0 + 16, k0, lane);
#pragma unroll
    for (int ni = 0; ni < 4; ++ni) {
      v16h b = frag_bt(w, kD, n0 + ni * 16, k0, lane);
      acc[0][ni] = WMMA_F16(a0, b, acc[0][ni]);
      acc[1][ni] = WMMA_F16(a1, b, acc[1][ni]);
    }
  }
  const int col = lane & 15;
  const int rbase = (lane >> 4) << 3;
  float bb[4];
#pragma unroll
  for (int ni = 0; ni < 4; ++ni) bb[ni] = bias[n0 + ni * 16 + col];
#pragma unroll
  for (int mi = 0; mi < 2; ++mi)
#pragma unroll
    for (int ni = 0; ni < 4; ++ni)
#pragma unroll
      for (int r = 0; r < 8; ++r) {
        float val = acc[mi][ni][r] + bb[ni];
        int row = m0 + mi * 16 + rbase + r;
        int n = n0 + ni * 16 + col;
        if (p == 0)      qh[(size_t)row * kD + n] = (half_t)val;
        else if (p == 1) kh[(size_t)row * kD + n] = (half_t)val;
        else {
          int bidx = row >> 11;          // row / 2048
          int t = row & (kS - 1);
          vT[((size_t)bidx * kD + n) * kS + t] = (half_t)val;  // V transposed
        }
      }
}

// -------------------------------------------------------- flash attention
// Workgroup: 16 query rows, full D=1024 output, 8 waves each own 16x128 of O.
__global__ __launch_bounds__(256) void attn_kernel(const half_t* __restrict__ qh,
                                                   const half_t* __restrict__ kh,
                                                   const half_t* __restrict__ vT,
                                                   float* __restrict__ out) {
  const int s0 = blockIdx.x * 16;
  const int b = blockIdx.y;
  const int tid = threadIdx.x;
  const int lane = tid & 31, wid = tid >> 5;
  const float scale = 0.03125f;  // 1/sqrt(1024)

  __shared__ __align__(16) half_t Qs[16 * 1024];   // 32 KB
  __shared__ __align__(16) half_t Ps[16 * 128];    // 4 KB
  __shared__ float red_max[8][16];
  __shared__ float red_sum[8][16];
  __shared__ float m_s[16], l_s[16], alpha_s[16], newm_s[16];

  const half_t* qbase = qh + (size_t)(b * kS + s0) * kD;
  const half_t* kbase = kh + (size_t)b * kS * kD;
  const half_t* vbase = vT + (size_t)b * kD * kS;

  for (int i = tid; i < (16 * 1024) / 8; i += 256)
    ((v8h*)Qs)[i] = ((const v8h*)qbase)[i];
  if (tid < 16) { m_s[tid] = -3.0e38f; l_s[tid] = 0.0f; }

  v8f acc[8] = {};
  const int rbase = (lane >> 4) << 3;
  const int col = lane & 15;

  __syncthreads();

  for (int t0 = 0; t0 < kS; t0 += 128) {
    // ---- scores: each wave computes one 16x16 tile, t = t0 + wid*16, K = D
    v8f sa = {};
    const int tw = t0 + wid * 16;
    for (int k0 = 0; k0 < kD; k0 += 32) {
      v16h aq = frag_a(Qs, 1024, 0, k0, lane);          // LDS
      v16h bk_ = frag_bt(kbase, kD, tw, k0, lane);      // K row-major == B^T
      sa = WMMA_F16(aq, bk_, sa);
    }
    float rv[8];
#pragma unroll
    for (int r = 0; r < 8; ++r) rv[r] = sa[r] * scale;
    // row max within the 16-lane group (xor of bits 0..3 stays in-group)
#pragma unroll
    for (int r = 0; r < 8; ++r) {
      float v = rv[r];
      v = fmaxf(v, __shfl_xor(v, 1));
      v = fmaxf(v, __shfl_xor(v, 2));
      v = fmaxf(v, __shfl_xor(v, 4));
      v = fmaxf(v, __shfl_xor(v, 8));
      if (col == 0) red_max[wid][rbase + r] = v;
    }
    __syncthreads();
    if (tid < 16) {
      float mx = m_s[tid];
#pragma unroll
      for (int w2 = 0; w2 < 8; ++w2) mx = fmaxf(mx, red_max[w2][tid]);
      newm_s[tid] = mx;
      alpha_s[tid] = __expf(m_s[tid] - mx);
    }
    __syncthreads();
    // ---- P = exp(s - m), write to LDS (f16), partial row sums
#pragma unroll
    for (int r = 0; r < 8; ++r) {
      float pval = __expf(rv[r] - newm_s[rbase + r]);
      Ps[(rbase + r) * 128 + wid * 16 + col] = (half_t)pval;
      float s = pval;
      s += __shfl_xor(s, 1);
      s += __shfl_xor(s, 2);
      s += __shfl_xor(s, 4);
      s += __shfl_xor(s, 8);
      if (col == 0) red_sum[wid][rbase + r] = s;
    }
    // ---- rescale running accumulators by alpha
    float al[8];
#pragma unroll
    for (int r = 0; r < 8; ++r) al[r] = alpha_s[rbase + r];
#pragma unroll
    for (int sub = 0; sub < 8; ++sub)
#pragma unroll
      for (int r = 0; r < 8; ++r) acc[sub][r] *= al[r];
    __syncthreads();
    if (tid < 16) {
      float s = 0.f;
#pragma unroll
      for (int w2 = 0; w2 < 8; ++w2) s += red_sum[w2][tid];
      l_s[tid] = l_s[tid] * alpha_s[tid] + s;
      m_s[tid] = newm_s[tid];
    }
    // ---- O += P @ V  (K = 128 local keys, wave's N slice = wid*128..+128)
#pragma unroll
    for (int k4 = 0; k4 < 4; ++k4) {
      v16h pa = frag_a(Ps, 128, 0, k4 * 32, lane);      // LDS
#pragma unroll
      for (int sub = 0; sub < 8; ++sub) {
        v16h vb = frag_bt(vbase, kS, wid * 128 + sub * 16, t0 + k4 * 32, lane);
        acc[sub] = WMMA_F16(pa, vb, acc[sub]);
      }
    }
  }

  __syncthreads();
  float linv[8];
#pragma unroll
  for (int r = 0; r < 8; ++r) linv[r] = 1.0f / l_s[rbase + r];
#pragma unroll
  for (int sub = 0; sub < 8; ++sub)
#pragma unroll
    for (int r = 0; r < 8; ++r)
      out[(size_t)(b * kS + s0 + rbase + r) * kD + wid * 128 + sub * 16 + col] =
          acc[sub][r] * linv[r];
}

// ---------------------------------------------------------------- launch
extern "C" void kernel_launch(void* const* d_in, const int* in_sizes, int n_in,
                              void* d_out, int out_size, void* d_ws, size_t ws_size,
                              hipStream_t stream) {
  (void)in_sizes; (void)n_in; (void)out_size; (void)ws_size;
  const float* x  = (const float*)d_in[0];
  const float* Wq = (const float*)d_in[1];
  const float* bq = (const float*)d_in[2];
  const float* Wk = (const float*)d_in[3];
  const float* bk = (const float*)d_in[4];
  const float* Wv = (const float*)d_in[5];
  const float* bv = (const float*)d_in[6];
  float* out = (float*)d_out;

  char* ws = (char*)d_ws;
  half_t* xh = (half_t*)(ws);                    // 16384*1024*2 = 32 MiB
  half_t* wT = (half_t*)(ws + 33554432);         // 3*1024*1024*2 = 6 MiB
  half_t* qh = (half_t*)(ws + 39845888);         // 32 MiB
  half_t* kh = (half_t*)(ws + 73400320);         // 32 MiB
  half_t* vT = (half_t*)(ws + 106954752);        // 32 MiB  (total ~134 MiB)

  cvt_x_kernel<<<dim3((kB * kS * kD) / (256 * 4)), 256, 0, stream>>>(x, xh);
  cvt_w_kernel<<<dim3(32, 32, 3), 256, 0, stream>>>(Wq, Wk, Wv, wT);
  qkv_kernel<<<dim3((kB * kS) / 128, kD / 128, 3), 256, 0, stream>>>(
      xh, wT, bq, bk, bv, qh, kh, vT);
  attn_kernel<<<dim3(kS / 16, kB), 256, 0, stream>>>(qh, kh, vT, out);
}
